// Hourglass_9380208574847
// MI455X (gfx1250) — matrix operand order repaired and checked by reading, not compiled, not run on hardware
//
#include <hip/hip_runtime.h>
#include <hip/hip_bf16.h>
#include <math.h>
#include <stdint.h>

// ---------------------------------------------------------------------------
// Types for CDNA5 WMMA (wave32): V_WMMA_F32_16X16X32_F16
// ---------------------------------------------------------------------------
typedef _Float16 h8   __attribute__((ext_vector_type(8)));
typedef _Float16 v16h __attribute__((ext_vector_type(16)));
typedef float    v8f  __attribute__((ext_vector_type(8)));

#define WMMA_F16(a, b, c) \
  __builtin_amdgcn_wmma_f32_16x16x32_f16(false, (a), false, (b), (short)0, (c), false, false)

#define T0     2048
#define C0     1024
#define H0     16
#define DH0    64
#define TP     512
#define CH     4096   // 4*C
#define NLAYER 6
#define C3     3072   // fused QKV width

static __device__ __forceinline__ v8f zero8() {
  v8f z;
#pragma unroll
  for (int i = 0; i < 8; ++i) z[i] = 0.0f;
  return z;
}

// A-fragment (and P-fragment) loader. k order used here (contiguous, matched by B loads):
// lanes 0-15 hold K = 0..15, lanes 16-31 hold K = 16..31.
// p points at row start of the 32-wide K tile; kb = 0 (lanes 0-15) or 8.
static __device__ __forceinline__ v16h lda16(const _Float16* p, int kb) {
  h8 lo = *(const h8*)(p + 2 * kb);
  h8 hi = *(const h8*)(p + 2 * kb + 8);
  v16h r;
#pragma unroll
  for (int i = 0; i < 8; ++i) { r[i] = lo[i]; r[i + 8] = hi[i]; }
  return r;
}

// ---------------------------------------------------------------------------
// Weight packing: fp32 -> f16, column-major per output (B operand layout):
// dst[l][n][k] contiguous in k, so a B fragment is one contiguous 16-half load
// (lanes 0-15: K=kt..kt+15, lanes 16-31: K=kt+16..kt+31).
// ---------------------------------------------------------------------------
__global__ void pack_rm_k(const float* __restrict__ src, _Float16* __restrict__ dst,
                          int Kd, int N, int total) {
  int i = blockIdx.x * blockDim.x + threadIdx.x;
  if (i >= total) return;
  int per = Kd * N;
  int l = i / per;
  int r = i - l * per;
  int n = r / Kd;
  int k = r - n * Kd;
  dst[i] = (_Float16)src[(size_t)l * per + (size_t)k * N + n];
}

// wq/wk/wv are (L, H, C, DH); pack into fused buffer:
// dst[l][sec*1024 + n=h*64+dh][k=c], per-layer stride 3M halves.
__global__ void pack_qkv_k(const float* __restrict__ src, _Float16* __restrict__ dst,
                           int sec, int total) {
  int i = blockIdx.x * blockDim.x + threadIdx.x;
  if (i >= total) return;
  int l = i >> 20;                 // per-layer source block = 1M elements
  int r = i & ((1 << 20) - 1);
  int n = r >> 10;                 // h*64+dh
  int c = r & 1023;
  int h = n >> 6, dh = n & 63;
  dst[(size_t)l * 3 * (1 << 20) + (size_t)sec * (1 << 20) + r] =
      (_Float16)src[((size_t)l << 20) + (size_t)(h * C0 + c) * DH0 + dh];
}

// ---------------------------------------------------------------------------
// LayerNorm: one 256-thread block per row (C0=1024). Writes f16 GEMM operand.
// ---------------------------------------------------------------------------
__global__ void __launch_bounds__(256) ln_k(const float* __restrict__ x,
                                            const float* __restrict__ g,
                                            const float* __restrict__ b,
                                            _Float16* __restrict__ out) {
  __shared__ float red[256];
  const int row = blockIdx.x, tid = threadIdx.x;
  const float* xr = x + (size_t)row * C0;
  float v[4];
  float s = 0.f;
#pragma unroll
  for (int i = 0; i < 4; ++i) { v[i] = xr[tid + 256 * i]; s += v[i]; }
  red[tid] = s; __syncthreads();
  for (int st = 128; st > 0; st >>= 1) { if (tid < st) red[tid] += red[tid + st]; __syncthreads(); }
  const float mu = red[0] * (1.0f / C0);
  __syncthreads();
  s = 0.f;
#pragma unroll
  for (int i = 0; i < 4; ++i) { float d = v[i] - mu; s += d * d; }
  red[tid] = s; __syncthreads();
  for (int st = 128; st > 0; st >>= 1) { if (tid < st) red[tid] += red[tid + st]; __syncthreads(); }
  const float rstd = rsqrtf(red[0] * (1.0f / C0) + 1e-5f);
  _Float16* orow = out + (size_t)row * C0;
#pragma unroll
  for (int i = 0; i < 4; ++i) {
    int c = tid + 256 * i;
    orow[c] = (_Float16)((v[i] - mu) * rstd * g[c] + b[c]);
  }
}

// ---------------------------------------------------------------------------
// WMMA GEMM: Y[M,N] = A[M,Kd] (f16 row-major) @ Bp (f16 packed [n][k])
// 8 waves/block, per-wave 64x64 tile -> block tile 128x256.
// 16 b128 loads : 16 WMMAs per K-step. Fused epilogue: +bias, optional ReLU,
// optional fp32 residual add, optional fp32 and/or f16 outputs.
// ---------------------------------------------------------------------------
__global__ void __launch_bounds__(256) gemm_k(const _Float16* __restrict__ A,
                                              const _Float16* __restrict__ Bp,
                                              const float* __restrict__ bias,
                                              const float* __restrict__ resid,
                                              float* __restrict__ outF,
                                              _Float16* __restrict__ outH,
                                              int M, int N, int Kd, int relu) {
  const int lane  = threadIdx.x & 31;
  const int wave  = threadIdx.x >> 5;
  const int wm    = wave >> 2, wn = wave & 3;
  const int mBase = blockIdx.y * 128 + wm * 64;
  const int nBase = blockIdx.x * 256 + wn * 64;
  const int lm    = lane & 15;
  const int kb    = (lane < 16) ? 0 : 8;
  const int bko   = (lane < 16) ? 0 : 16;
  const int rbase = (lane < 16) ? 0 : 8;

  const _Float16* ap[4];
#pragma unroll
  for (int i = 0; i < 4; ++i) ap[i] = A + (size_t)(mBase + i * 16 + lm) * Kd;
  const _Float16* bptr[4];
#pragma unroll
  for (int j = 0; j < 4; ++j) bptr[j] = Bp + (size_t)(nBase + j * 16 + lm) * Kd + bko;

  v8f acc[4][4];
#pragma unroll
  for (int i = 0; i < 4; ++i)
#pragma unroll
    for (int j = 0; j < 4; ++j) acc[i][j] = zero8();

  for (int kt = 0; kt < Kd; kt += 32) {
    v16h a[4];
#pragma unroll
    for (int i = 0; i < 4; ++i) a[i] = lda16(ap[i] + kt, kb);
#pragma unroll
    for (int j = 0; j < 4; ++j) {
      v16h b = *(const v16h*)(bptr[j] + kt);
#pragma unroll
      for (int i = 0; i < 4; ++i) acc[i][j] = WMMA_F16(a[i], b, acc[i][j]);
    }
  }

#pragma unroll
  for (int i = 0; i < 4; ++i) {
#pragma unroll
    for (int j = 0; j < 4; ++j) {
      const int col = nBase + j * 16 + lm;
      const float bi = bias ? bias[col] : 0.0f;
#pragma unroll
      for (int r = 0; r < 8; ++r) {
        const size_t idx = (size_t)(mBase + i * 16 + rbase + r) * N + col;
        float v = acc[i][j][r] + bi;
        if (relu) v = fmaxf(v, 0.0f);
        if (resid) v += resid[idx];
        if (outF) outF[idx] = v;
        if (outH) outH[idx] = (_Float16)v;
      }
    }
  }
}

// ---------------------------------------------------------------------------
// Flash attention (causal): grid (H0, T/256), 8 waves/block, 32 queries/wave.
// Q/K/V share a fused [t][3072] f16 buffer (ld = 3072); O is [t][1024] f16.
// S = Q K^T via WMMA with the K tile staged block-wide into LDS by the
// gfx1250 async-LDS path (global_load_async_to_lds_b128 / ASYNCcnt);
// online softmax through padded LDS; P.V via WMMA with V transposed in LDS.
// ---------------------------------------------------------------------------
__global__ void __launch_bounds__(256) flash_k(const _Float16* __restrict__ Qg,
                                               const _Float16* __restrict__ Kg,
                                               const _Float16* __restrict__ Vg,
                                               int ld,
                                               _Float16* __restrict__ Og,
                                               float scale) {
  __shared__ _Float16 Kl[32][64];        // K tile, row-major == B-frag layout
  __shared__ _Float16 Vt[64][32];        // V tile transposed: [dh][key]
  __shared__ float    Ssh[8][32][33];    // raw logits per wave (padded)
  __shared__ _Float16 Psh[8][32][40];    // probs per wave (16B-aligned rows)
  __shared__ float    scl[8][32];        // per-row rescale factors
  __shared__ float    lsh[8][32];        // per-row softmax denominators

  const int h     = blockIdx.x;
  const int qb    = blockIdx.y;
  const int wave  = threadIdx.x >> 5;
  const int lane  = threadIdx.x & 31;
  const int lm    = lane & 15;
  const int kb    = (lane < 16) ? 0 : 8;
  const int bko   = (lane < 16) ? 0 : 16;
  const int rbase = (lane < 16) ? 0 : 8;
  const int hoff  = h * DH0;
  const int qw0   = qb * 256 + wave * 32;

  // Per-thread assignment for the async K-tile stage: 256 threads x 16B = 4KB.
  const int kkey = threadIdx.x >> 3;          // 0..31
  const int kseg = (threadIdx.x & 7) * 8;     // 0,8,..,56 (halves)
  const uint32_t klds = (uint32_t)(uintptr_t)(&Kl[kkey][kseg]);

  // Preload Q fragments: 2 query sub-tiles x 2 K-steps over DH=64.
  v16h qf[2][2];
#pragma unroll
  for (int i = 0; i < 2; ++i)
#pragma unroll
    for (int ks = 0; ks < 2; ++ks)
      qf[i][ks] = lda16(Qg + (size_t)(qw0 + i * 16 + lm) * ld + hoff + ks * 32, kb);

  v8f o[2][4];
#pragma unroll
  for (int i = 0; i < 2; ++i)
#pragma unroll
    for (int j = 0; j < 4; ++j) o[i][j] = zero8();

  float mrun = -1e30f, lrun = 0.f;
  const int nkt   = qb * 8 + 8;     // key tiles up to block's last query
  const int lastq = qw0 + 31;

  for (int kt = 0; kt < nkt; ++kt) {
    // Async stage of the whole K tile into LDS (one b128 per lane).
    {
      const _Float16* kg = Kg + (size_t)(kt * 32 + kkey) * ld + hoff + kseg;
      asm volatile("global_load_async_to_lds_b128 %0, %1, off"
                   :: "v"(klds), "v"(kg) : "memory");
    }
    // Cooperative V-tile transpose into LDS (32 keys x 64 dh).
#pragma unroll
    for (int e = 0; e < 8; ++e) {
      int idx = threadIdx.x * 8 + e;
      int key = idx >> 6, dh = idx & 63;
      Vt[dh][key] = Vg[(size_t)(kt * 32 + key) * ld + hoff + dh];
    }
    asm volatile("s_wait_asynccnt 0" ::: "memory");
    __syncthreads();

    // Wave-uniform predicate (keep EXEC all-ones around WMMA).
    const int active = __builtin_amdgcn_readfirstlane((kt * 32 <= lastq) ? 1 : 0);
    if (active) {
      v8f sacc[2][2];
#pragma unroll
      for (int i = 0; i < 2; ++i)
#pragma unroll
        for (int j = 0; j < 2; ++j) sacc[i][j] = zero8();
#pragma unroll
      for (int ks = 0; ks < 2; ++ks) {
#pragma unroll
        for (int j = 0; j < 2; ++j) {
          v16h bf = *(const v16h*)(&Kl[j * 16 + lm][ks * 32 + bko]);
          sacc[0][j] = WMMA_F16(qf[0][ks], bf, sacc[0][j]);
          sacc[1][j] = WMMA_F16(qf[1][ks], bf, sacc[1][j]);
        }
      }
#pragma unroll
      for (int i = 0; i < 2; ++i)
#pragma unroll
        for (int j = 0; j < 2; ++j)
#pragma unroll
          for (int r = 0; r < 8; ++r)
            Ssh[wave][i * 16 + rbase + r][j * 16 + lm] = sacc[i][j][r];
    }
    __syncthreads();

    if (active) {
      // Lane owns query row = lane within the wave's 32-query strip.
      const float* srow = Ssh[wave][lane];
      const int q = qw0 + lane;
      float mtile = -1e30f;
      for (int j = 0; j < 32; ++j) {
        float v = srow[j] * scale;
        if (kt * 32 + j > q) v = -1e30f;
        mtile = fmaxf(mtile, v);
      }
      const float mnew = fmaxf(mrun, mtile);
      float lsum = 0.f;
      for (int j = 0; j < 32; ++j) {
        float v = srow[j] * scale;
        if (kt * 32 + j > q) v = -1e30f;
        float p = __expf(v - mnew);
        lsum += p;
        Psh[wave][lane][j] = (_Float16)p;
      }
      const float sf = __expf(mrun - mnew);
      lrun = lrun * sf + lsum;
      mrun = mnew;
      scl[wave][lane] = sf;
    }
    __syncthreads();

    if (active) {
      v16h pf0 = lda16(&Psh[wave][lm][0], kb);
      v16h pf1 = lda16(&Psh[wave][16 + lm][0], kb);
#pragma unroll
      for (int i = 0; i < 2; ++i)
#pragma unroll
        for (int r = 0; r < 8; ++r) {
          const float f = scl[wave][i * 16 + rbase + r];
#pragma unroll
          for (int j = 0; j < 4; ++j) o[i][j][r] *= f;
        }
#pragma unroll
      for (int j = 0; j < 4; ++j) {
        v16h vf = *(const v16h*)(&Vt[j * 16 + lm][bko]);
        o[0][j] = WMMA_F16(pf0, vf, o[0][j]);
        o[1][j] = WMMA_F16(pf1, vf, o[1][j]);
      }
    }
    __syncthreads();   // protect Kl/Vt before the next tile overwrites them
  }

  lsh[wave][lane] = lrun;
  __syncthreads();
#pragma unroll
  for (int i = 0; i < 2; ++i)
#pragma unroll
    for (int r = 0; r < 8; ++r) {
      const int qrow = i * 16 + rbase + r;
      const float inv = 1.0f / lsh[wave][qrow];
#pragma unroll
      for (int j = 0; j < 4; ++j)
        Og[(size_t)(qw0 + qrow) * C0 + hoff + j * 16 + lm] =
            (_Float16)(o[i][j][r] * inv);
    }
}

// ---------------------------------------------------------------------------
// Pool gather: xs = shift-right-3(x); out[t'][kk*C+cc] = f16(xs[4t'+kk][cc])
// ---------------------------------------------------------------------------
__global__ void pool_gather_k(const float* __restrict__ x, _Float16* __restrict__ out,
                              int total) {
  int i = blockIdx.x * blockDim.x + threadIdx.x;
  if (i >= total) return;
  int tp = i >> 12;
  int rem = i & 4095;
  int kk = rem >> 10;
  int cc = rem & 1023;
  int t = tp * 4 + kk;
  float v = (t < 3) ? 0.f : x[(size_t)(t - 3) * C0 + cc];
  out[i] = (_Float16)v;
}

// x[t] += xp[t/4]  (repeat_interleave upsample + residual)
__global__ void upsample_add_k(float* __restrict__ x, const float* __restrict__ xp,
                               int total) {
  int i = blockIdx.x * blockDim.x + threadIdx.x;
  if (i >= total) return;
  int t = i / C0, c = i - t * C0;
  x[i] += xp[(size_t)(t >> 2) * C0 + c];
}

// ---------------------------------------------------------------------------
// Host driver
// ---------------------------------------------------------------------------
static void launch_gemm(const _Float16* A, const _Float16* Bp, const float* bias,
                        const float* resid, float* outF, _Float16* outH,
                        int M, int N, int Kd, int relu, hipStream_t s) {
  dim3 grid(N / 256, M / 128);
  gemm_k<<<grid, 256, 0, s>>>(A, Bp, bias, resid, outF, outH, M, N, Kd, relu);
}

extern "C" void kernel_launch(void* const* d_in, const int* in_sizes, int n_in,
                              void* d_out, int out_size, void* d_ws, size_t ws_size,
                              hipStream_t stream) {
  const float* x_in  = (const float*)d_in[0];
  const float* ln1_g = (const float*)d_in[1];
  const float* ln1_b = (const float*)d_in[2];
  const float* ln2_g = (const float*)d_in[3];
  const float* ln2_b = (const float*)d_in[4];
  const float* wq    = (const float*)d_in[5];
  const float* wk    = (const float*)d_in[6];
  const float* wv    = (const float*)d_in[7];
  const float* wo    = (const float*)d_in[8];
  const float* bo    = (const float*)d_in[9];
  const float* w1    = (const float*)d_in[10];
  const float* b1    = (const float*)d_in[11];
  const float* w2    = (const float*)d_in[12];
  const float* b2    = (const float*)d_in[13];
  const float* projw = (const float*)d_in[14];
  const float* projb = (const float*)d_in[15];

  // Workspace layout (~205 MB), 256B-aligned chunks.
  char* base = (char*)d_ws;
  size_t off = 0;
  auto take = [&](size_t bytes) -> char* {
    char* p = base + off;
    off = (off + bytes + 255) & ~(size_t)255;
    return p;
  };
  _Float16* qkvp   = (_Float16*)take((size_t)NLAYER * C0 * C3 * 2);  // fused Q|K|V
  _Float16* wop    = (_Float16*)take((size_t)NLAYER * C0 * C0 * 2);
  _Float16* w1p    = (_Float16*)take((size_t)NLAYER * C0 * CH * 2);
  _Float16* w2p    = (_Float16*)take((size_t)NLAYER * C0 * CH * 2);
  _Float16* projp  = (_Float16*)take((size_t)CH * C0 * 2);
  float*    xbuf   = (float*)take((size_t)T0 * C0 * 4);
  float*    xp     = (float*)take((size_t)TP * C0 * 4);
  _Float16* h16    = (_Float16*)take((size_t)T0 * C0 * 2);
  _Float16* qkv16  = (_Float16*)take((size_t)T0 * C3 * 2);           // fused Q|K|V acts
  _Float16* o16    = (_Float16*)take((size_t)T0 * C0 * 2);
  _Float16* mh16   = (_Float16*)take((size_t)T0 * CH * 2);
  _Float16* xpin16 = (_Float16*)take((size_t)TP * CH * 2);
  (void)ws_size; (void)in_sizes; (void)n_in; (void)out_size;

  // ---- Pack weights fp32 -> f16 WMMA layouts (every call: deterministic) ----
  {
    int tq = NLAYER * C0 * C0;
    pack_qkv_k<<<(tq + 255) / 256, 256, 0, stream>>>(wq, qkvp, 0, tq);
    pack_qkv_k<<<(tq + 255) / 256, 256, 0, stream>>>(wk, qkvp, 1, tq);
    pack_qkv_k<<<(tq + 255) / 256, 256, 0, stream>>>(wv, qkvp, 2, tq);
    pack_rm_k<<<(tq + 255) / 256, 256, 0, stream>>>(wo, wop, C0, C0, tq);
    int t1 = NLAYER * C0 * CH;
    pack_rm_k<<<(t1 + 255) / 256, 256, 0, stream>>>(w1, w1p, C0, CH, t1);
    pack_rm_k<<<(t1 + 255) / 256, 256, 0, stream>>>(w2, w2p, CH, C0, t1);
    int tp2 = CH * C0;
    pack_rm_k<<<(tp2 + 255) / 256, 256, 0, stream>>>(projw, projp, CH, C0, tp2);
  }

  hipMemcpyAsync(xbuf, x_in, (size_t)T0 * C0 * 4, hipMemcpyDeviceToDevice, stream);

  const float att_scale = 0.125f;  // 1/sqrt(64)

  auto run_layer = [&](int l, float* xb, int T) {
    // ln1 -> h16 ; fused QKV GEMM ; flash attention ; Wo + bo + residual -> x
    ln_k<<<T, 256, 0, stream>>>(xb, ln1_g + (size_t)l * C0, ln1_b + (size_t)l * C0, h16);
    launch_gemm(h16, qkvp + (size_t)l * C0 * C3, nullptr, nullptr, nullptr, qkv16,
                T, C3, C0, 0, stream);
    flash_k<<<dim3(H0, T / 256), 256, 0, stream>>>(qkv16, qkv16 + C0, qkv16 + 2 * C0,
                                                   C3, o16, att_scale);
    launch_gemm(o16, wop + (size_t)l * C0 * C0, bo + (size_t)l * C0, xb, xb, nullptr,
                T, C0, C0, 0, stream);
    // ln2 -> h16 ; MLP: relu(h@w1+b1)@w2+b2 + residual -> x
    ln_k<<<T, 256, 0, stream>>>(xb, ln2_g + (size_t)l * C0, ln2_b + (size_t)l * C0, h16);
    launch_gemm(h16, w1p + (size_t)l * C0 * CH, b1 + (size_t)l * CH, nullptr, nullptr,
                mh16, T, CH, C0, 1, stream);
    launch_gemm(mh16, w2p + (size_t)l * C0 * CH, b2 + (size_t)l * C0, xb, xb, nullptr,
                T, C0, CH, 0, stream);
  };

  // Outer pre-layers at T=2048
  run_layer(0, xbuf, T0);
  run_layer(1, xbuf, T0);

  // Shift-right(3) + 4x pooling projection -> xp (512 x 1024)
  {
    int tot = TP * CH;
    pool_gather_k<<<(tot + 255) / 256, 256, 0, stream>>>(xbuf, xpin16, tot);
    launch_gemm(xpin16, projp, projb, nullptr, xp, nullptr, TP, C0, CH, 0, stream);
  }

  // Inner layers at T=512
  run_layer(2, xp, TP);
  run_layer(3, xp, TP);

  // Upsample (repeat_interleave by 4) + residual
  {
    int tot = T0 * C0;
    upsample_add_k<<<(tot + 255) / 256, 256, 0, stream>>>(xbuf, xp, tot);
  }

  // Outer post-layers at T=2048
  run_layer(4, xbuf, T0);
  run_layer(5, xbuf, T0);

  hipMemcpyAsync(d_out, xbuf, (size_t)T0 * C0 * 4, hipMemcpyDeviceToDevice, stream);
}
